// ExHuneLSTM_51256139710949
// MI455X (gfx1250) — matrix operand order repaired and checked by reading, not compiled, not run on hardware
//
#include <hip/hip_runtime.h>
#include <math.h>

#define B_    32
#define T_    1024
#define D_    512
#define H_    512
#define NWG   32
#define NTHR  128

typedef unsigned short u16_t;
typedef __attribute__((ext_vector_type(16))) __bf16 v16bf;
typedef __attribute__((ext_vector_type(8)))  float  v8f;

union BV { uint4 q[2]; v16bf v; };

__device__ __forceinline__ u16_t f2bf(float f) {
  union { float f; unsigned u; } a; a.f = f;
  unsigned u = a.u;
  unsigned r = (u + 0x7FFFu + ((u >> 16) & 1u)) >> 16;
  return (u16_t)r;
}
__device__ __forceinline__ float bf2f(u16_t b) {
  union { unsigned u; float f; } a; a.u = ((unsigned)b) << 16;
  return a.f;
}
__device__ __forceinline__ float sigm(float x)  { return 1.0f / (1.0f + __expf(-x)); }
__device__ __forceinline__ float tanhfast(float x) { return 2.0f / (1.0f + __expf(-2.0f * x)) - 1.0f; }

// ---------------- init: convert u -> bf16, zero state/barrier -----------------
__global__ void init_kernel(const float* __restrict__ u, u16_t* __restrict__ u16,
                            u16_t* __restrict__ hbufs, unsigned* __restrict__ bar,
                            float* __restrict__ xhat) {
  long n = (long)B_ * T_ * D_;
  long i = (long)blockIdx.x * blockDim.x + threadIdx.x;
  long stride = (long)gridDim.x * blockDim.x;
  for (long e = i; e < n; e += stride) u16[e] = f2bf(u[e]);
  long hn = 4L * B_ * H_;
  for (long e = i; e < hn; e += stride) hbufs[e] = 0;
  if (i == 0) *bar = 0u;
  if (i < B_) xhat[i] = 0.0f;
}

// ---------------- persistent recurrent kernel ---------------------------------
__global__ void __launch_bounds__(NTHR)
rnn_kernel(const float* __restrict__ u,
           const float* __restrict__ wih_f, const float* __restrict__ whh_f,
           const float* __restrict__ bih_f, const float* __restrict__ bhh_f,
           const float* __restrict__ wih_b, const float* __restrict__ whh_b,
           const float* __restrict__ bih_b, const float* __restrict__ bhh_b,
           const u16_t* __restrict__ u16,
           u16_t* __restrict__ hbufs,      // [h_f | h_b | h1_f | h1_b], each B_*H_
           float* __restrict__ xhat_buf,   // B_ floats
           unsigned* __restrict__ bar,
           float* __restrict__ out)
{
  const int tid  = threadIdx.x;
  const int wg   = blockIdx.x;
  const int cell = wg & 1;          // 0 = forward cell, 1 = "backward" cell
  const int jj   = wg >> 1;         // column-pair group (0..15)
  const int ch   = tid >> 6;        // column group within the pair (0/1)
  const int mt   = (tid >> 5) & 1;  // M tile (0/1)
  const int lane = tid & 31;
  const int nl   = lane & 15;
  const int hi   = lane >> 4;
  const int n0   = jj * 32 + ch * 16;
  const int mbase = mt * 16 + 8 * hi;

  extern __shared__ u16_t smem[];
  u16_t* sWih = smem;               // [128 rows][512] bf16, row = ch*64 + q*16 + r
  u16_t* sWhh = smem + 128 * 512;

  const float* Wih = cell ? wih_b : wih_f;
  const float* Whh = cell ? whh_b : whh_f;
  const float* Bih = cell ? bih_b : bih_f;
  const float* Bhh = cell ? bhh_b : bhh_f;

  // --- stage this workgroup's 128 Wih rows + 128 Whh rows into LDS (bf16) ---
  {
    int chf = tid >> 6, q = (tid >> 4) & 3, r = tid & 15;
    int row = q * H_ + jj * 32 + chf * 16 + r;
    int rl  = chf * 64 + q * 16 + r;
    const float* s0 = Wih + (long)row * D_;
    u16_t* d0 = sWih + rl * D_;
    for (int k = 0; k < D_; ++k) d0[k] = f2bf(s0[k]);
    const float* s1 = Whh + (long)row * H_;
    u16_t* d1 = sWhh + rl * D_;
    for (int k = 0; k < H_; ++k) d1[k] = f2bf(s1[k]);
  }
  __syncthreads();

  // --- per-lane constants: fused bias and Wih row-sums (rank-1 trick) ---
  float bias[4], rs[4];
#pragma unroll
  for (int q = 0; q < 4; ++q) {
    int row = q * H_ + n0 + nl;
    bias[q] = Bih[row] + Bhh[row];
    const float* src = Wih + (long)row * D_;
    float s = 0.0f;
    for (int k = 0; k < D_; ++k) s += src[k];
    rs[q] = s;
  }

  u16_t* h_f  = hbufs;
  u16_t* h_b  = hbufs + 1 * B_ * H_;
  u16_t* h1_f = hbufs + 2 * B_ * H_;
  u16_t* h1_b = hbufs + 3 * B_ * H_;
  u16_t* h_live = cell ? h_b  : h_f;
  u16_t* h1     = cell ? h1_b : h1_f;

  v8f cstate = {};                  // fp32 cell state, resident in VGPRs
  v8f xg[4];                        // cached u_t @ Wih^T tiles (reused phase 2)
  unsigned round_ = 0;

#define GRID_BARRIER()                                                        \
  do {                                                                        \
    __syncthreads();                                                          \
    if (tid == 0) {                                                           \
      __threadfence();                                                        \
      atomicAdd(bar, 1u);                                                     \
      ++round_;                                                               \
      unsigned tgt = round_ * (unsigned)NWG;                                  \
      while (__hip_atomic_load(bar, __ATOMIC_RELAXED,                         \
                               __HIP_MEMORY_SCOPE_AGENT) < tgt)               \
        __builtin_amdgcn_s_sleep(1);                                          \
      __threadfence();                                                        \
    }                                                                         \
    __syncthreads();                                                          \
  } while (0)

  // load A tile + 4 B tiles for one k-step into the given buffer set
#define LOADAB(AROW, SW, K0, A_, B4_)                                         \
  do {                                                                        \
    (A_).q[0] = *(const uint4*)((AROW) + (K0) + 8 * hi);                      \
    (A_).q[1] = *(const uint4*)((AROW) + (K0) + 16 + 8 * hi);                 \
    _Pragma("unroll")                                                         \
    for (int q = 0; q < 4; ++q) {                                             \
      const u16_t* bp = (SW) + (ch * 64 + q * 16 + nl) * D_ + (K0);           \
      (B4_)[q].q[0] = *(const uint4*)(bp + 8 * hi);                           \
      (B4_)[q].q[1] = *(const uint4*)(bp + 16 + 8 * hi);                      \
    }                                                                         \
  } while (0)

#define WMMA4(A_, B4_, ACC)                                                   \
  do {                                                                        \
    _Pragma("unroll")                                                         \
    for (int q = 0; q < 4; ++q)                                               \
      ACC[q] = __builtin_amdgcn_wmma_f32_16x16x32_bf16(                       \
                 false, (A_).v, false, (B4_)[q].v, (short)0, ACC[q],          \
                 false, false);                                               \
  } while (0)

  // software-pipelined 16-step GEMM: prefetch ks+1 operands before WMMAs of ks
#define GEMM16(AROW, SW, ACC)                                                 \
  do {                                                                        \
    BV a0_, a1_, b0_[4], b1_[4];                                              \
    LOADAB(AROW, SW, 0, a0_, b0_);                                            \
    _Pragma("unroll")                                                         \
    for (int ks = 0; ks < 16; ks += 2) {                                      \
      if (ks + 1 < 16) LOADAB(AROW, SW, (ks + 1) * 32, a1_, b1_);             \
      WMMA4(a0_, b0_, ACC);                                                   \
      if (ks + 2 < 16) LOADAB(AROW, SW, (ks + 2) * 32, a0_, b0_);             \
      WMMA4(a1_, b1_, ACC);                                                   \
    }                                                                         \
  } while (0)

  for (int t = 0; t < T_; ++t) {
    // ---------------- phase 1 window ----------------
    // write out[t-1] = 1.5*u + 0.5*x_hat + 0.5*x_hat_next (producers complete
    // behind barrier B of t-1; overwriters run after barrier A of t)
    if (t > 0) {
      int e0 = (wg * NTHR + tid) * 4;
      int b = e0 >> 9, d0 = e0 & 511;
      float add = 0.5f * xhat_buf[b] + 0.5f * bf2f(h_b[b * H_ + (H_ - 1)]);
      const float* up = u   + ((long)b * T_ + (t - 1)) * D_ + d0;
      float*       op = out + ((long)b * T_ + (t - 1)) * D_ + d0;
#pragma unroll
      for (int i = 0; i < 4; ++i) op[i] = 1.5f * up[i] + add;
    }

    // x-GEMM: xg[q] = u_t @ Wih^T  (this wave's 16 rows x 16 cols per gate)
#pragma unroll
    for (int q = 0; q < 4; ++q) xg[q] = (v8f){};
    {
      const u16_t* arow = u16 + ((long)(mt * 16 + nl) * T_ + t) * D_;
      GEMM16(arow, sWih, xg);
    }

    // h-GEMM 1: g[q] = xg[q] + h_live @ Whh^T
    v8f g[4];
#pragma unroll
    for (int q = 0; q < 4; ++q) g[q] = xg[q];
    {
      const u16_t* arow = h_live + (mt * 16 + nl) * H_;
      GEMM16(arow, sWhh, g);
    }

    // LSTM update 1 -> h1, c (c stays in registers)
#pragma unroll
    for (int v = 0; v < 8; ++v) {
      float gi = g[0][v] + bias[0];
      float gf = g[1][v] + bias[1];
      float gg = g[2][v] + bias[2];
      float go = g[3][v] + bias[3];
      float c2 = sigm(gf) * cstate[v] + sigm(gi) * tanhfast(gg);
      float h2 = sigm(go) * tanhfast(c2);
      cstate[v] = c2;
      h1[(mbase + v) * H_ + n0 + nl] = f2bf(h2);
    }

    GRID_BARRIER();   // barrier A: h1_f/h1_b complete everywhere

    // ---------------- phase 2 window ----------------
    // x_hat[m] = h1_b[m, H-1]; publish to xhat_buf for the out[t] write
    float xh[8];
#pragma unroll
    for (int v = 0; v < 8; ++v) xh[v] = bf2f(h1_b[(mbase + v) * H_ + (H_ - 1)]);
    if (wg == 0 && tid < B_) xhat_buf[tid] = bf2f(h1_b[tid * H_ + (H_ - 1)]);

    // rank-1 corrected input gates: g = xg + x_hat[m]*rowsum(Wih)[n]
#pragma unroll
    for (int q = 0; q < 4; ++q) {
      v8f tv;
#pragma unroll
      for (int v = 0; v < 8; ++v) tv[v] = xg[q][v] + xh[v] * rs[q];
      g[q] = tv;
    }

    // h-GEMM 2: g[q] += h1 @ Whh^T
    {
      const u16_t* arow = h1 + (mt * 16 + nl) * H_;
      GEMM16(arow, sWhh, g);
    }

    // LSTM update 2 -> h_live (the new live state), c in registers
#pragma unroll
    for (int v = 0; v < 8; ++v) {
      float gi = g[0][v] + bias[0];
      float gf = g[1][v] + bias[1];
      float gg = g[2][v] + bias[2];
      float go = g[3][v] + bias[3];
      float c2 = sigm(gf) * cstate[v] + sigm(gi) * tanhfast(gg);
      float h2 = sigm(go) * tanhfast(c2);
      cstate[v] = c2;
      h_live[(mbase + v) * H_ + n0 + nl] = f2bf(h2);
    }

    GRID_BARRIER();   // barrier B: h_f/h_b (live) + xhat_buf complete
  }

  // final output row t = T-1
  {
    int e0 = (wg * NTHR + tid) * 4;
    int b = e0 >> 9, d0 = e0 & 511;
    float add = 0.5f * xhat_buf[b] + 0.5f * bf2f(h_b[b * H_ + (H_ - 1)]);
    const float* up = u   + ((long)b * T_ + (T_ - 1)) * D_ + d0;
    float*       op = out + ((long)b * T_ + (T_ - 1)) * D_ + d0;
#pragma unroll
    for (int i = 0; i < 4; ++i) op[i] = 1.5f * up[i] + add;
  }
#undef GEMM16
#undef WMMA4
#undef LOADAB
#undef GRID_BARRIER
}

extern "C" void kernel_launch(void* const* d_in, const int* in_sizes, int n_in,
                              void* d_out, int out_size, void* d_ws, size_t ws_size,
                              hipStream_t stream) {
  const float* u     = (const float*)d_in[0];
  const float* wih_f = (const float*)d_in[1];
  const float* whh_f = (const float*)d_in[2];
  const float* bih_f = (const float*)d_in[3];
  const float* bhh_f = (const float*)d_in[4];
  const float* wih_b = (const float*)d_in[5];
  const float* whh_b = (const float*)d_in[6];
  const float* bih_b = (const float*)d_in[7];
  const float* bhh_b = (const float*)d_in[8];
  float* out = (float*)d_out;

  // workspace layout
  u16_t* u16   = (u16_t*)d_ws;                       // B*T*D bf16 = 32 MB
  u16_t* hbufs = u16 + (size_t)B_ * T_ * D_;         // 4 * B*H bf16 = 128 KB
  float* xhat  = (float*)(hbufs + 4 * B_ * H_);      // 32 floats
  unsigned* bar = (unsigned*)(xhat + B_);            // 1 uint

  init_kernel<<<256, 256, 0, stream>>>(u, u16, hbufs, bar, xhat);

  size_t lds_bytes = (size_t)2 * 128 * 512 * sizeof(u16_t);  // 256 KB
  rnn_kernel<<<NWG, NTHR, lds_bytes, stream>>>(
      u, wih_f, whh_f, bih_f, bhh_f, wih_b, whh_b, bih_b, bhh_b,
      u16, hbufs, xhat, bar, out);
}
